// SelfAttention_7851200217994
// MI455X (gfx1250) — matrix operand order repaired and checked by reading, not compiled, not run on hardware
//
#include <hip/hip_runtime.h>
#include <hip/hip_bf16.h>
#include <math.h>

// ---------------------------------------------------------------------------
// Sizes (reference: B=4, S=2048, D=1024)
// ---------------------------------------------------------------------------
#define BATCH   4
#define SEQ     2048
#define DIM     1024
#define MROWS   (BATCH * SEQ)     // 8192
#define FFT_N   1024
#define FFT_LOG 10
#define FREQ    513               // rfft bins 0..512
#define FPAD    520               // padded complex stride (float2 units)
#define CHUNKS  16                // cumsum chunks along S
#define CSIZE   (SEQ / CHUNKS)    // 128

// GEMM tiling
#define BM 64
#define BN 64
#define BK 32
#define LSTR 40                   // LDS row stride in halfs (80B: 16B aligned)

typedef __attribute__((ext_vector_type(16))) _Float16 v16h;
typedef __attribute__((ext_vector_type(8)))  _Float16 v8h;
typedef __attribute__((ext_vector_type(4)))  _Float16 v4h;
typedef __attribute__((ext_vector_type(8)))  float    v8f;
typedef __attribute__((ext_vector_type(4)))  float    v4f;

union F16x16 { v16h v; v8h h[2]; };

// ---------------------------------------------------------------------------
// Pass 0a: convert x to f16 (vectorized x4)
// ---------------------------------------------------------------------------
__global__ __launch_bounds__(256) void cvt_x_kernel(
    const float* __restrict__ src, _Float16* __restrict__ dst, int n4) {
  int i = blockIdx.x * blockDim.x + threadIdx.x;
  if (i < n4) {
    v4f s = ((const v4f*)src)[i];
    v4h h;
    #pragma unroll
    for (int j = 0; j < 4; ++j) h[j] = (_Float16)s[j];
    ((v4h*)dst)[i] = h;
  }
}

// ---------------------------------------------------------------------------
// Pass 0b: convert the three weight matrices to f16 (packed [3][D][D])
// ---------------------------------------------------------------------------
__global__ __launch_bounds__(256) void cvt_w_kernel(
    const float* __restrict__ wq, const float* __restrict__ wk,
    const float* __restrict__ wv, _Float16* __restrict__ w16, int n4) {
  int i = blockIdx.x * blockDim.x + threadIdx.x;
  if (i >= n4) return;
  const float* srcs[3] = {wq, wk, wv};
  #pragma unroll
  for (int mat = 0; mat < 3; ++mat) {
    v4f s = ((const v4f*)srcs[mat])[i];
    v4h h;
    #pragma unroll
    for (int j = 0; j < 4; ++j) h[j] = (_Float16)s[j];
    ((v4h*)(w16 + (size_t)mat * n4 * 4))[i] = h;
  }
}

// ---------------------------------------------------------------------------
// Pass 1: fused QKV GEMM with WMMA + double-buffered ASYNC global->LDS copies.
//   A = x16 [M x K] f16 ;  B = W^T, i.e. B[k=d][n=e] = W[e][d]
//   out[mat][m][n] = sum_d x[m][d] * W[mat][n][d]
// 8 waves: wave = (nsel<<2)|mband ; each wave: 1 M-tile x 2 N-tiles x 3 mats.
// Staging uses GLOBAL_LOAD_ASYNC_TO_LDS_B128 (ASYNCcnt) so global latency of
// stage i+1 overlaps the WMMA work of stage i. All six B fragments are
// preloaded into distinct registers so the 12 ds_load_b128 issue as one
// clause and the six WMMAs retire back-to-back.
// ---------------------------------------------------------------------------
__global__ __launch_bounds__(256) void qkv_gemm_kernel(
    const _Float16* __restrict__ x16,   // [M][K] f16
    const _Float16* __restrict__ w16,   // [3][N][K] f16
    float* __restrict__ qkv,            // [3][M][N] f32
    int M, int K, int N) {
  __shared__ __align__(16) _Float16 sh_x[2][BM * LSTR];
  __shared__ __align__(16) _Float16 sh_w[2][3][BN * LSTR];

  const int tid   = threadIdx.x;
  const int lane  = tid & 31;
  const int wave  = tid >> 5;       // 0..7
  const int mband = wave & 3;       // 16-row band within BM
  const int nsel  = wave >> 2;      // which 32-col half of BN
  const int m0    = blockIdx.y * BM;
  const int n0    = blockIdx.x * BN;

  const int srow = tid >> 2;        // 0..63 staging row
  const int scol = (tid & 3) * 8;   // 0,8,16,24 staging col base

  // per-thread global staging bases (advance by k0 halfs each stage)
  const _Float16* gx  = x16 + (size_t)(m0 + srow) * K + scol;
  const _Float16* gw0 = w16 + (size_t)(n0 + srow) * K + scol;
  const _Float16* gw1 = gw0 + (size_t)N * K;
  const _Float16* gw2 = gw1 + (size_t)N * K;

  // issue 4 async 16B copies (x + 3 W tiles) for K-chunk k0 into buffer b
  auto stage = [&](int k0, int b) {
    const unsigned lx  = (unsigned)(uintptr_t)&sh_x[b][srow * LSTR + scol];
    const unsigned lw0 = (unsigned)(uintptr_t)&sh_w[b][0][srow * LSTR + scol];
    const unsigned lw1 = (unsigned)(uintptr_t)&sh_w[b][1][srow * LSTR + scol];
    const unsigned lw2 = (unsigned)(uintptr_t)&sh_w[b][2][srow * LSTR + scol];
    asm volatile("global_load_async_to_lds_b128 %0, %1, off"
                 :: "v"(lx), "v"(gx + k0) : "memory");
    asm volatile("global_load_async_to_lds_b128 %0, %1, off"
                 :: "v"(lw0), "v"(gw0 + k0) : "memory");
    asm volatile("global_load_async_to_lds_b128 %0, %1, off"
                 :: "v"(lw1), "v"(gw1 + k0) : "memory");
    asm volatile("global_load_async_to_lds_b128 %0, %1, off"
                 :: "v"(lw2), "v"(gw2 + k0) : "memory");
  };

  const v8f vzero = {0.f, 0.f, 0.f, 0.f, 0.f, 0.f, 0.f, 0.f};
  v8f acc[3][2];
  #pragma unroll
  for (int mat = 0; mat < 3; ++mat)
    #pragma unroll
    for (int nt = 0; nt < 2; ++nt) acc[mat][nt] = vzero;

  stage(0, 0);                       // prologue: fill buffer 0

  const int nsteps = K / BK;
  for (int it = 0; it < nsteps; ++it) {
    const int cur = it & 1;
    if (it + 1 < nsteps) {
      stage((it + 1) * BK, cur ^ 1); // stream next tile while computing
      if (it + 2 < nsteps)           // warm L2 for the stage after next
        __builtin_prefetch(gx + (it + 2) * BK, 0, 1);
      // 4 outstanding belong to the just-issued stage; older 4 are done
      asm volatile("s_wait_asynccnt 0x4" ::: "memory");
    } else {
      asm volatile("s_wait_asynccnt 0x0" ::: "memory");
    }
    __syncthreads();                 // all waves' cur-buffer data visible

    // ---- A fragment: 16-bit 16x32 layout ----
    // lane<16: K in {kb0..kb0+7} u {kb0+16..kb0+23}, kb0=0 ; lane>=16: kb0=8
    const _Float16* bx = sh_x[cur];
    F16x16 a;
    {
      const int m   = mband * 16 + (lane & 15);
      const int kb0 = (lane < 16) ? 0 : 8;
      a.h[0] = *(const v8h*)(&bx[m * LSTR + kb0]);
      a.h[1] = *(const v8h*)(&bx[m * LSTR + kb0 + 16]);
    }
    // ---- preload all six B fragments (distinct registers) ----
    // lane<16: K in {0..7} u {16..23} of column n ; lane>=16: {8..15} u {24..31}
    const int kb = (lane < 16) ? 0 : 8;
    F16x16 bfr[2][3];
    #pragma unroll
    for (int nt = 0; nt < 2; ++nt) {
      const int n = nsel * 32 + nt * 16 + (lane & 15);
      #pragma unroll
      for (int mat = 0; mat < 3; ++mat) {
        const v8h* p = (const v8h*)(&sh_w[cur][mat][n * LSTR + kb]);
        bfr[nt][mat].h[0] = p[0];
        bfr[nt][mat].h[1] = p[2];
      }
    }
    // ---- WMMA burst ----
    #pragma unroll
    for (int nt = 0; nt < 2; ++nt)
      #pragma unroll
      for (int mat = 0; mat < 3; ++mat)
        acc[mat][nt] = __builtin_amdgcn_wmma_f32_16x16x32_f16(
            false, a.v, false, bfr[nt][mat].v, (short)0, acc[mat][nt],
            false, false);
    __syncthreads();                 // done reading cur before it is reused
  }

  // ---- epilogue: C layout VGPR r -> M=r (lanes 0-15) / M=r+8 (lanes 16-31)
  #pragma unroll
  for (int mat = 0; mat < 3; ++mat) {
    float* outp = qkv + (size_t)mat * M * N;
    #pragma unroll
    for (int nt = 0; nt < 2; ++nt) {
      const int nn = n0 + nsel * 32 + nt * 16 + (lane & 15);
      #pragma unroll
      for (int r = 0; r < 8; ++r) {
        const int mm = m0 + mband * 16 + ((lane < 16) ? r : r + 8);
        outp[(size_t)mm * N + nn] = acc[mat][nt][r];
      }
    }
  }
}

// ---------------------------------------------------------------------------
// Shared-memory radix-2 complex FFT of length 1024, 256 threads.
// sign = +1.0f : forward (e^{-2pi i nk/N}) ; sign = -1.0f : inverse (unscaled)
// ---------------------------------------------------------------------------
__device__ inline void fft1024_lds(float* re, float* im, int tid, float sign) {
  // bit-reverse permutation (each (i,j) pair swapped exactly once)
  for (int i = tid; i < FFT_N; i += 256) {
    int j = (int)(__brev((unsigned)i) >> (32 - FFT_LOG));
    if (j > i) {
      float tr = re[i]; re[i] = re[j]; re[j] = tr;
      float ti = im[i]; im[i] = im[j]; im[j] = ti;
    }
  }
  __syncthreads();
  for (int stage = 0; stage < FFT_LOG; ++stage) {
    const int half = 1 << stage;
    for (int b = tid; b < FFT_N / 2; b += 256) {
      const int grp = b >> stage;
      const int pos = b & (half - 1);
      const int i0  = (grp << (stage + 1)) + pos;
      const int i1  = i0 + half;
      const float ang = sign * (-6.28318530717958647692f) *
                        (float)pos / (float)(half * 2);
      const float wr = __cosf(ang), wi = __sinf(ang);
      const float ar = re[i0], ai = im[i0];
      const float br = re[i1], bi = im[i1];
      const float tr = br * wr - bi * wi;
      const float ti = br * wi + bi * wr;
      re[i0] = ar + tr; im[i0] = ai + ti;
      re[i1] = ar - tr; im[i1] = ai - ti;
    }
    __syncthreads();
  }
}

// ---------------------------------------------------------------------------
// Pass 2: per row -> Fk=rfft(k), phase-normalize; Fv=rfft(v); Fkv = Fku*Fv
// ---------------------------------------------------------------------------
__global__ __launch_bounds__(256) void kv_fft_kernel(
    const float* __restrict__ kmat, const float* __restrict__ vmat,
    float2* __restrict__ Fkv) {
  __shared__ float re[FFT_N], im[FFT_N];
  __shared__ float kur[FREQ], kui[FREQ];
  const int row = blockIdx.x;
  const int tid = threadIdx.x;

  // ---- k ----
  const float* kp = kmat + (size_t)row * DIM;
  for (int i = tid; i < FFT_N; i += 256) { re[i] = kp[i]; im[i] = 0.f; }
  __syncthreads();
  fft1024_lds(re, im, tid, 1.f);
  for (int f = tid; f < FREQ; f += 256) {
    const float a = re[f], b = im[f];
    const float m = sqrtf(a * a + b * b) + 1e-8f;
    kur[f] = a / m; kui[f] = b / m;
  }
  __syncthreads();

  // ---- v ----
  const float* vp = vmat + (size_t)row * DIM;
  for (int i = tid; i < FFT_N; i += 256) { re[i] = vp[i]; im[i] = 0.f; }
  __syncthreads();
  fft1024_lds(re, im, tid, 1.f);
  for (int f = tid; f < FREQ; f += 256) {
    const float a = re[f], b = im[f];           // Fv
    float2 o;
    o.x = kur[f] * a - kui[f] * b;              // Fku * Fv
    o.y = kur[f] * b + kui[f] * a;
    Fkv[(size_t)row * FPAD + f] = o;
  }
}

// ---------------------------------------------------------------------------
// Pass 3: causal cumsum of Fkv over S, per (batch, frequency). 3-phase scan.
// ---------------------------------------------------------------------------
__global__ __launch_bounds__(256) void cumsum_local_kernel(
    float2* __restrict__ Fkv, float2* __restrict__ totals) {
  const int f = blockIdx.x * blockDim.x + threadIdx.x;
  if (f >= FREQ) return;
  const int b  = blockIdx.y / CHUNKS;
  const int ch = blockIdx.y % CHUNKS;
  size_t base = ((size_t)b * SEQ + (size_t)ch * CSIZE) * FPAD + f;
  float2 acc; acc.x = 0.f; acc.y = 0.f;
  for (int s = 0; s < CSIZE; ++s) {
    float2 t = Fkv[base + (size_t)s * FPAD];
    acc.x += t.x; acc.y += t.y;
    Fkv[base + (size_t)s * FPAD] = acc;
  }
  totals[((size_t)b * CHUNKS + ch) * FPAD + f] = acc;
}

__global__ __launch_bounds__(256) void scan_totals_kernel(
    float2* __restrict__ totals) {
  const int f = blockIdx.x * blockDim.x + threadIdx.x;
  if (f >= FREQ) return;
  const int b = blockIdx.y;
  float2 run; run.x = 0.f; run.y = 0.f;
  for (int ch = 0; ch < CHUNKS; ++ch) {
    const size_t idx = ((size_t)b * CHUNKS + ch) * FPAD + f;
    float2 t = totals[idx];
    totals[idx] = run;                          // exclusive prefix
    run.x += t.x; run.y += t.y;
  }
}

__global__ __launch_bounds__(256) void cumsum_add_kernel(
    float2* __restrict__ Fkv, const float2* __restrict__ totals) {
  const int f = blockIdx.x * blockDim.x + threadIdx.x;
  if (f >= FREQ) return;
  const int b  = blockIdx.y / CHUNKS;
  const int ch = blockIdx.y % CHUNKS;
  const float2 off = totals[((size_t)b * CHUNKS + ch) * FPAD + f];
  size_t base = ((size_t)b * SEQ + (size_t)ch * CSIZE) * FPAD + f;
  for (int s = 0; s < CSIZE; ++s) {
    float2 t = Fkv[base + (size_t)s * FPAD];
    t.x += off.x; t.y += off.y;
    Fkv[base + (size_t)s * FPAD] = t;
  }
}

// ---------------------------------------------------------------------------
// Pass 4: per row -> Fq=rfft(q), phase-normalize; G = Fkv_cum * conj(Fqu);
// rebuild full spectrum by conjugate symmetry; inverse FFT; scale 1/N.
// ---------------------------------------------------------------------------
__global__ __launch_bounds__(256) void unbind_ifft_kernel(
    const float* __restrict__ qmat, const float2* __restrict__ Fkvc,
    float* __restrict__ out) {
  __shared__ float re[FFT_N], im[FFT_N];
  __shared__ float qur[FREQ], qui[FREQ];
  const int row = blockIdx.x;
  const int tid = threadIdx.x;

  const float* qp = qmat + (size_t)row * DIM;
  for (int i = tid; i < FFT_N; i += 256) { re[i] = qp[i]; im[i] = 0.f; }
  __syncthreads();
  fft1024_lds(re, im, tid, 1.f);
  for (int f = tid; f < FREQ; f += 256) {
    const float a = re[f], b = im[f];
    const float m = sqrtf(a * a + b * b) + 1e-8f;
    qur[f] = a / m; qui[f] = b / m;
  }
  __syncthreads();

  for (int f = tid; f < FREQ; f += 256) {
    const float2 s = Fkvc[(size_t)row * FPAD + f];
    const float gr = s.x * qur[f] + s.y * qui[f];   // S * conj(Qu)
    const float gi = s.y * qur[f] - s.x * qui[f];
    re[f] = gr; im[f] = gi;
    if (f >= 1 && f < FFT_N / 2) {                  // conjugate symmetry
      re[FFT_N - f] = gr; im[FFT_N - f] = -gi;
    }
  }
  __syncthreads();
  fft1024_lds(re, im, tid, -1.f);
  const float invN = 1.f / (float)FFT_N;
  for (int i = tid; i < FFT_N; i += 256)
    out[(size_t)row * DIM + i] = re[i] * invN;
}

// ---------------------------------------------------------------------------
// Launch
// ---------------------------------------------------------------------------
extern "C" void kernel_launch(void* const* d_in, const int* in_sizes, int n_in,
                              void* d_out, int out_size, void* d_ws,
                              size_t ws_size, hipStream_t stream) {
  (void)in_sizes; (void)n_in; (void)out_size; (void)ws_size;
  const float* x  = (const float*)d_in[0];   // [B,S,D] f32
  const float* Wq = (const float*)d_in[1];   // [D,D]
  const float* Wk = (const float*)d_in[2];
  const float* Wv = (const float*)d_in[3];
  float* out = (float*)d_out;                // [B,S,D] f32

  const int M = MROWS, K = DIM, N = DIM;

  // workspace carve-up (256B aligned)
  char* base = (char*)d_ws;
  size_t off = 0;
  auto carve = [&](size_t bytes) -> char* {
    char* p = base + off;
    off = (off + bytes + 255) & ~(size_t)255;
    return p;
  };
  _Float16* x16   = (_Float16*)carve((size_t)M * K * sizeof(_Float16));
  _Float16* w16   = (_Float16*)carve((size_t)3 * N * K * sizeof(_Float16));
  float*    qkv   = (float*)   carve((size_t)3 * M * N * sizeof(float));
  float2*   Fkv   = (float2*)  carve((size_t)M * FPAD * sizeof(float2));
  float2*   tot   = (float2*)  carve((size_t)BATCH * CHUNKS * FPAD * sizeof(float2));

  // 0) f32 -> f16 conversions (x and the three weight matrices)
  cvt_x_kernel<<<dim3((M * K / 4 + 255) / 256), dim3(256), 0, stream>>>(
      x, x16, M * K / 4);
  cvt_w_kernel<<<dim3((N * K / 4 + 255) / 256), dim3(256), 0, stream>>>(
      Wq, Wk, Wv, w16, N * K / 4);

  // 1) fused QKV projection (WMMA + async global->LDS double buffering)
  qkv_gemm_kernel<<<dim3(N / BN, M / BM), dim3(256), 0, stream>>>(
      x16, w16, qkv, M, K, N);

  const float* q = qkv;
  const float* k = qkv + (size_t)M * N;
  const float* v = qkv + (size_t)2 * M * N;

  // 2) bind in frequency domain: Fkv = unit(rfft(k)) * rfft(v)
  kv_fft_kernel<<<dim3(M), dim3(256), 0, stream>>>(k, v, Fkv);

  // 3) causal cumsum over S (per batch & frequency), chunked 3-phase scan
  const dim3 fgrid((FREQ + 255) / 256, BATCH * CHUNKS);
  cumsum_local_kernel<<<fgrid, dim3(256), 0, stream>>>(Fkv, tot);
  scan_totals_kernel<<<dim3((FREQ + 255) / 256, BATCH), dim3(256), 0, stream>>>(tot);
  cumsum_add_kernel<<<fgrid, dim3(256), 0, stream>>>(Fkv, tot);

  // 4) unbind with unit(rfft(q)) and inverse FFT to time domain
  unbind_ifft_kernel<<<dim3(M), dim3(256), 0, stream>>>(q, Fkv, out);
}
